// GeomExtendedContrastiveLoss_3891240370580
// MI455X (gfx1250) — matrix operand order repaired and checked by reading, not compiled, not run on hardware
//
#include <hip/hip_runtime.h>
#include <hip/hip_bf16.h>
#include <math.h>

// Problem constants (fixed by the reference: B=4, N=2048, D=128)
#define B_   4
#define N_   2048
#define D_   128
#define TM   64          // workgroup row/col tile
#define ZS   136         // LDS stride (ushorts) for zn tiles (128 + pad, 16B-aligned rows)
#define APS  72          // LDS stride (ushorts) for adj tiles (64 + pad, 16B-aligned rows)
#define EPS_ 1e-9f
#define TAU_INV 14.285714285714286f   // 1/0.07

typedef __attribute__((ext_vector_type(16))) __bf16 v16bf;
typedef __attribute__((ext_vector_type(8)))  __bf16 v8bf;
typedef __attribute__((ext_vector_type(8)))  float  v8f;

__device__ __forceinline__ unsigned short f2bf_bits(float f) {
  union { float f; unsigned u; } v; v.f = f;
  unsigned u = v.u;
  u += 0x7FFFu + ((u >> 16) & 1u);          // round-to-nearest-even
  return (unsigned short)(u >> 16);
}

__device__ __forceinline__ v8f vzero8() {
  v8f z;
  for (int i = 0; i < 8; ++i) z[i] = 0.0f;
  return z;
}

__device__ __forceinline__ v8f wmma_bf16(v16bf a, v16bf b, v8f c) {
  // v_wmma_f32_16x16x32_bf16  (neg_a, A, neg_b, B, c_mod, C, reuse_a, reuse_b)
  return __builtin_amdgcn_wmma_f32_16x16x32_bf16(false, a, false, b, (short)0, c,
                                                 false, false);
}

// A operand: 16x32 bf16. ISA layout: lane<16 -> row M=lane, K 0-7 & 16-23;
// lane>=16 -> row M=lane-16, K 8-15 & 24-31.
__device__ __forceinline__ v16bf ldsA(const unsigned short* lds, int row0, int k0, int stride) {
  const int lane = threadIdx.x & 31;
  const int m = lane & 15, hi = lane >> 4;
  const unsigned short* p = lds + (size_t)(row0 + m) * stride + k0 + hi * 8;
  v8bf x0 = *(const v8bf*)(p);        // K k0+{0..7} or {8..15}
  v8bf x1 = *(const v8bf*)(p + 16);   // K k0+{16..23} or {24..31}
  return __builtin_shufflevector(x0, x1, 0,1,2,3,4,5,6,7,8,9,10,11,12,13,14,15);
}

// B operand: 32x16 bf16. ISA layout: lane<16 -> col N=lane, K 0-7 & 16-23;
// lane>=16 -> col N=lane-16, K 8-15 & 24-31. Tile stored as rows of Y (cols of C).
__device__ __forceinline__ v16bf ldsB(const unsigned short* lds, int col0, int k0, int stride) {
  const int lane = threadIdx.x & 31;
  const int n = lane & 15, hi = lane >> 4;
  const unsigned short* p = lds + (size_t)(col0 + n) * stride + k0 + hi * 8;
  v8bf x0 = *(const v8bf*)(p);
  v8bf x1 = *(const v8bf*)(p + 16);
  return __builtin_shufflevector(x0, x1, 0,1,2,3,4,5,6,7,8,9,10,11,12,13,14,15);
}

// Copy 64 contiguous rows x 128 bf16 (global, row-major) into LDS tile of stride ZS.
__device__ __forceinline__ void stage_zn(unsigned short* dst, const unsigned short* src) {
  const uint4* s = (const uint4*)src;              // 8 bf16 per uint4
  for (int e = threadIdx.x; e < 64 * 16; e += (int)blockDim.x) {
    int row = e >> 4, q = e & 15;
    *(uint4*)(dst + (size_t)row * ZS + q * 8) = s[e];
  }
}

// ---------------------------------------------------------------- row inverse
__global__ void k_rowinv(const float* __restrict__ adj, float* __restrict__ invr) {
  const int row = blockIdx.x;                      // 0 .. B*N-1
  const float* p = adj + (size_t)row * N_;
  float s = 0.f;
  for (int c = threadIdx.x; c < N_; c += 256) s += p[c];
  for (int m = 16; m >= 1; m >>= 1) s += __shfl_xor(s, m, 32);
  __shared__ float sm[8];
  const int lane = threadIdx.x & 31, wv = threadIdx.x >> 5;
  if (lane == 0) sm[wv] = s;
  __syncthreads();
  if (threadIdx.x == 0) {
    float t = 0.f;
    for (int i = 0; i < 8; ++i) t += sm[i];
    invr[row] = 1.0f / (t + EPS_);
  }
}

// ------------------------------------------------------------- l2norm -> bf16
__global__ void k_norm(const float* __restrict__ z, unsigned short* __restrict__ zn) {
  const int row  = blockIdx.x * 8 + (threadIdx.x >> 5);   // one wave per row
  const int lane = threadIdx.x & 31;
  const float* p = z + (size_t)row * D_;
  float a0 = p[lane], a1 = p[lane + 32], a2 = p[lane + 64], a3 = p[lane + 96];
  float ss = a0 * a0 + a1 * a1 + a2 * a2 + a3 * a3;
  for (int m = 16; m >= 1; m >>= 1) ss += __shfl_xor(ss, m, 32);
  float inv = 1.0f / fmaxf(sqrtf(ss), 1e-12f);
  unsigned short* o = zn + (size_t)row * D_;
  o[lane]      = f2bf_bits(a0 * inv);
  o[lane + 32] = f2bf_bits(a1 * inv);
  o[lane + 64] = f2bf_bits(a2 * inv);
  o[lane + 96] = f2bf_bits(a3 * inv);
}

// ------------------------------------------- preconvert A' and adj^T to bf16
// One 64x64 tile per block. Abf[i][k] = adj[i][k]*invr[k]; Tbf[j][k] = adj[k][j].
__global__ __launch_bounds__(256) void k_prep(const float* __restrict__ adj,
                                              const float* __restrict__ invr,
                                              unsigned short* __restrict__ Abf,
                                              unsigned short* __restrict__ Tbf) {
  __shared__ float t[64][65];
  const int blk = blockIdx.x;
  const int b  = blk >> 10;                 // 1024 tiles per batch (32x32)
  const int tr = (blk >> 5) & 31, tc = blk & 31;
  const int r0 = tr * 64, c0 = tc * 64;
  const float* adjb  = adj  + (size_t)b * N_ * N_;
  const float* invrb = invr + (size_t)b * N_;
  unsigned short* Ab = Abf + (size_t)b * N_ * N_;
  unsigned short* Tb = Tbf + (size_t)b * N_ * N_;
  for (int e = threadIdx.x; e < 64 * 64; e += 256) {
    int r = e >> 6, c = e & 63;
    float a = adjb[(size_t)(r0 + r) * N_ + (c0 + c)];
    Ab[(size_t)(r0 + r) * N_ + (c0 + c)] = f2bf_bits(a * invrb[c0 + c]);
    t[r][c] = a;
  }
  __syncthreads();
  for (int e = threadIdx.x; e < 64 * 64; e += 256) {
    int c = e >> 6, r = e & 63;             // write row j=c0+c, cols k=r0+r (coalesced)
    Tb[(size_t)(c0 + c) * N_ + (r0 + r)] = f2bf_bits(t[r][c]);
  }
}

// ----------------------------------------------- score = aG + (1-a)S, top-1
// grid.x = B * (N/TM); block = 128 (4 waves, each owns 16 full rows)
// PRE=true : A'/adj^T pre-converted to bf16 in ws, 64-wide k-steps, pure B128 staging
// PRE=false: fallback, inline f32->bf16 conversion during staging (32-wide k-steps)
template <bool PRE>
__global__ __launch_bounds__(128) void k_score(const float* __restrict__ adj,
                                               const float* __restrict__ invr,
                                               const unsigned short* __restrict__ zn,
                                               const unsigned short* __restrict__ Abf,
                                               const unsigned short* __restrict__ Tbf,
                                               float* __restrict__ wout,
                                               int* __restrict__ iout) {
  __shared__ __align__(16) unsigned short Azn[64 * ZS];
  __shared__ __align__(16) unsigned short Bzn[64 * ZS];
  __shared__ __align__(16) unsigned short Ap [64 * APS];
  __shared__ __align__(16) unsigned short Bp [64 * APS];

  const int tid = threadIdx.x, lane = tid & 31, wv = tid >> 5;
  const int hi = lane >> 4, ln = lane & 15;
  const int b  = blockIdx.x / (N_ / TM);
  const int i0 = (blockIdx.x % (N_ / TM)) * TM;
  const float* adjb  = adj  + (size_t)b * N_ * N_;
  const float* invrb = invr + (size_t)b * N_;
  const unsigned short* znb = zn + (size_t)b * N_ * D_;
  const unsigned short* Ab = PRE ? Abf + (size_t)b * N_ * N_ : (const unsigned short*)0;
  const unsigned short* Tb = PRE ? Tbf + (size_t)b * N_ * N_ : (const unsigned short*)0;

  stage_zn(Azn, znb + (size_t)i0 * D_);
  __syncthreads();

  // ---- pass A: softmax row sums at the 3 feature scales
  float s0[8] = {}, s1[8] = {}, s2[8] = {};
  for (int j0 = 0; j0 < N_; j0 += TM) {
    __syncthreads();
    stage_zn(Bzn, znb + (size_t)j0 * D_);
    __syncthreads();
    v8f c[4];
#pragma unroll
    for (int t = 0; t < 4; ++t) c[t] = vzero8();
#pragma unroll
    for (int k0 = 0; k0 < D_; k0 += 32) {
      v16bf a = ldsA(Azn, wv * 16, k0, ZS);
#pragma unroll
      for (int tn = 0; tn < 4; ++tn) c[tn] = wmma_bf16(a, ldsB(Bzn, tn * 16, k0, ZS), c[tn]);
    }
#pragma unroll
    for (int tn = 0; tn < 4; ++tn)
#pragma unroll
      for (int v = 0; v < 8; ++v) {
        float s = c[tn][v];
        s0[v] += expf(s * 20.0f);
        s1[v] += expf(s * 10.0f);
        s2[v] += expf(s * 5.0f);
      }
  }
  // butterfly over the 16-lane column group -> every lane holds full row sums
#pragma unroll
  for (int m = 1; m < 16; m <<= 1)
#pragma unroll
    for (int v = 0; v < 8; ++v) {
      s0[v] += __shfl_xor(s0[v], m, 32);
      s1[v] += __shfl_xor(s1[v], m, 32);
      s2[v] += __shfl_xor(s2[v], m, 32);
    }
  float is0[8], is1[8], is2[8], iri[8];
#pragma unroll
  for (int v = 0; v < 8; ++v) {
    is0[v] = 1.0f / (3.0f * s0[v]);
    is1[v] = 1.0f / (3.0f * s1[v]);
    is2[v] = 1.0f / (3.0f * s2[v]);
    iri[v] = invrb[i0 + wv * 16 + v + 8 * hi];
  }

  // ---- pass B: score + argmax (streams P@P tiles fused with sim tiles)
  float best[8]; int bidx[8];
#pragma unroll
  for (int v = 0; v < 8; ++v) { best[v] = -1e30f; bidx[v] = 0; }

  for (int j0 = 0; j0 < N_; j0 += TM) {
    __syncthreads();
    stage_zn(Bzn, znb + (size_t)j0 * D_);
    __syncthreads();
    v8f sc[4];
#pragma unroll
    for (int t = 0; t < 4; ++t) sc[t] = vzero8();
#pragma unroll
    for (int k0 = 0; k0 < D_; k0 += 32) {
      v16bf a = ldsA(Azn, wv * 16, k0, ZS);
#pragma unroll
      for (int tn = 0; tn < 4; ++tn) sc[tn] = wmma_bf16(a, ldsB(Bzn, tn * 16, k0, ZS), sc[tn]);
    }
    v8f g[4];
#pragma unroll
    for (int t = 0; t < 4; ++t) g[t] = vzero8();

    if (PRE) {
      // 64-wide k-steps, pure 128-bit bf16 copies, no conversion in the hot loop
      for (int k0 = 0; k0 < N_; k0 += 64) {
        __syncthreads();
        for (int e = tid; e < 64 * 8; e += 128) {
          int row = e >> 3, q = e & 7;
          *(uint4*)(Ap + row * APS + q * 8) =
              *(const uint4*)(Ab + (size_t)(i0 + row) * N_ + k0 + q * 8);
          *(uint4*)(Bp + row * APS + q * 8) =
              *(const uint4*)(Tb + (size_t)(j0 + row) * N_ + k0 + q * 8);
        }
        __syncthreads();
#pragma unroll
        for (int ks = 0; ks < 64; ks += 32) {
          v16bf a = ldsA(Ap, wv * 16, ks, APS);
#pragma unroll
          for (int tn = 0; tn < 4; ++tn)
            g[tn] = wmma_bf16(a, ldsB(Bp, tn * 16, ks, APS), g[tn]);
        }
      }
    } else {
      // fallback: stage + convert from f32 adj on the fly
      for (int k0 = 0; k0 < N_; k0 += 32) {
        __syncthreads();
        for (int e = tid; e < 64 * 32; e += 128) {
          int rr = e >> 5, cc = e & 31;
          Ap[rr * APS + cc] =
              f2bf_bits(adjb[(size_t)(i0 + rr) * N_ + (k0 + cc)] * invrb[k0 + cc]);
        }
        for (int e = tid; e < 32 * 64; e += 128) {
          int kk = e >> 6, jj = e & 63;
          Bp[jj * APS + kk] = f2bf_bits(adjb[(size_t)(k0 + kk) * N_ + (j0 + jj)]);
        }
        __syncthreads();
        v16bf a = ldsA(Ap, wv * 16, 0, APS);
#pragma unroll
        for (int tn = 0; tn < 4; ++tn)
          g[tn] = wmma_bf16(a, ldsB(Bp, tn * 16, 0, APS), g[tn]);
      }
    }

    // combine into score, keep running top-1 (tie -> lowest j, as jax top_k)
#pragma unroll
    for (int tn = 0; tn < 4; ++tn)
#pragma unroll
      for (int v = 0; v < 8; ++v) {
        int i = i0 + wv * 16 + v + 8 * hi;
        int j = j0 + tn * 16 + ln;
        float aij = adjb[(size_t)i * N_ + j];
        float G = 0.5f * iri[v] * (aij + g[tn][v]);       // (P + P@P)/2
        float s = sc[tn][v];
        float S = expf(s * 20.0f) * is0[v] + expf(s * 10.0f) * is1[v] +
                  expf(s * 5.0f) * is2[v];
        float score = 0.5f * G + 0.5f * S;                // ALPHA = 0.5
        if (score > best[v] || (score == best[v] && j < bidx[v])) {
          best[v] = score; bidx[v] = j;
        }
      }
  }
  // reduce top-1 across the 16-lane column group
#pragma unroll
  for (int m = 1; m < 16; m <<= 1)
#pragma unroll
    for (int v = 0; v < 8; ++v) {
      float ov = __shfl_xor(best[v], m, 32);
      int   oi = __shfl_xor(bidx[v], m, 32);
      if (ov > best[v] || (ov == best[v] && oi < bidx[v])) { best[v] = ov; bidx[v] = oi; }
    }
  if (ln == 0) {
#pragma unroll
    for (int v = 0; v < 8; ++v) {
      int i = i0 + wv * 16 + v + 8 * hi;
      wout[(size_t)b * N_ + i] = best[v];
      iout[(size_t)b * N_ + i] = bidx[v];
    }
  }
}

// ------------------------------------------------------- contrastive loss rows
// grid.x = B * (2N/TM); block = 128
__global__ __launch_bounds__(128) void k_loss(const unsigned short* __restrict__ zn1,
                                              const unsigned short* __restrict__ zn2,
                                              const float* __restrict__ w1,
                                              const int* __restrict__ idx1,
                                              const float* __restrict__ w2,
                                              const int* __restrict__ idx2,
                                              float* __restrict__ lossbuf) {
  __shared__ __align__(16) unsigned short Azn[64 * ZS];
  __shared__ __align__(16) unsigned short Bzn[64 * ZS];
  const int tid = threadIdx.x, lane = tid & 31, wv = tid >> 5;
  const int hi = lane >> 4, ln = lane & 15;
  const int tiles = (2 * N_) / TM;
  const int b  = blockIdx.x / tiles;
  const int i0 = (blockIdx.x % tiles) * TM;
  const unsigned short* z1b = zn1 + (size_t)b * N_ * D_;
  const unsigned short* z2b = zn2 + (size_t)b * N_ * D_;

  stage_zn(Azn, (i0 < N_) ? z1b + (size_t)i0 * D_ : z2b + (size_t)(i0 - N_) * D_);
  __syncthreads();

  int strongc[8], gcol[8]; float wgt[8];
#pragma unroll
  for (int v = 0; v < 8; ++v) {
    int i = i0 + wv * 16 + v + 8 * hi;
    if (i < N_) {
      strongc[v] = N_ + i;
      int t = idx2[(size_t)b * N_ + i];
      wgt[v] = w2[(size_t)b * N_ + i];
      gcol[v] = (t == i) ? -1 : (N_ + t);
    } else {
      int ii = i - N_;
      strongc[v] = ii;
      int t = idx1[(size_t)b * N_ + ii];
      wgt[v] = w1[(size_t)b * N_ + ii];
      gcol[v] = (t == ii) ? -1 : t;
    }
  }

  float denom[8] = {}, extra[8] = {};
  for (int j0 = 0; j0 < 2 * N_; j0 += TM) {
    __syncthreads();
    stage_zn(Bzn, (j0 < N_) ? z1b + (size_t)j0 * D_ : z2b + (size_t)(j0 - N_) * D_);
    __syncthreads();
    v8f c[4];
#pragma unroll
    for (int t = 0; t < 4; ++t) c[t] = vzero8();
#pragma unroll
    for (int k0 = 0; k0 < D_; k0 += 32) {
      v16bf a = ldsA(Azn, wv * 16, k0, ZS);
#pragma unroll
      for (int tn = 0; tn < 4; ++tn) c[tn] = wmma_bf16(a, ldsB(Bzn, tn * 16, k0, ZS), c[tn]);
    }
#pragma unroll
    for (int tn = 0; tn < 4; ++tn)
#pragma unroll
      for (int v = 0; v < 8; ++v) {
        int i = i0 + wv * 16 + v + 8 * hi;
        int j = j0 + tn * 16 + ln;
        float e = expf(c[tn][v] * TAU_INV);
        if (j != i)          denom[v] += e;        // exact diagonal exclusion
        if (j == strongc[v]) extra[v] += e;
        if (j == gcol[v])    extra[v] += wgt[v] * e;
      }
  }
#pragma unroll
  for (int m = 1; m < 16; m <<= 1)
#pragma unroll
    for (int v = 0; v < 8; ++v) {
      denom[v] += __shfl_xor(denom[v], m, 32);
      extra[v] += __shfl_xor(extra[v], m, 32);
    }
  if (ln == 0) {
#pragma unroll
    for (int v = 0; v < 8; ++v) {
      int i = i0 + wv * 16 + v + 8 * hi;
      lossbuf[(size_t)b * (2 * N_) + i] = -logf((extra[v] + EPS_) / (denom[v] + EPS_));
    }
  }
}

// ------------------------------------------------------------- final mean
__global__ void k_reduce(const float* __restrict__ lossbuf, float* __restrict__ out) {
  __shared__ float sm[256];
  const int n = B_ * 2 * N_;
  float s = 0.f;
  for (int i = threadIdx.x; i < n; i += 256) s += lossbuf[i];
  sm[threadIdx.x] = s;
  __syncthreads();
  for (int off = 128; off > 0; off >>= 1) {
    if (threadIdx.x < off) sm[threadIdx.x] += sm[threadIdx.x + off];
    __syncthreads();
  }
  if (threadIdx.x == 0) out[0] = sm[0] / (float)n;
}

// ---------------------------------------------------------------- launcher
extern "C" void kernel_launch(void* const* d_in, const int* in_sizes, int n_in,
                              void* d_out, int out_size, void* d_ws, size_t ws_size,
                              hipStream_t stream) {
  (void)in_sizes; (void)n_in; (void)out_size;
  const float* z1   = (const float*)d_in[0];
  const float* z2   = (const float*)d_in[1];
  const float* adj1 = (const float*)d_in[2];
  const float* adj2 = (const float*)d_in[3];

  char* w = (char*)d_ws;
  float* invr1 = (float*)w;          w += (size_t)B_ * N_ * 4;
  float* invr2 = (float*)w;          w += (size_t)B_ * N_ * 4;
  unsigned short* zn1 = (unsigned short*)w; w += (size_t)B_ * N_ * D_ * 2;
  unsigned short* zn2 = (unsigned short*)w; w += (size_t)B_ * N_ * D_ * 2;
  float* w1 = (float*)w;             w += (size_t)B_ * N_ * 4;
  float* w2 = (float*)w;             w += (size_t)B_ * N_ * 4;
  int*   i1 = (int*)w;               w += (size_t)B_ * N_ * 4;
  int*   i2 = (int*)w;               w += (size_t)B_ * N_ * 4;
  float* lossbuf = (float*)w;        w += (size_t)B_ * 2 * N_ * 4;
  // optional bf16 operand buffers (reused for adj1 then adj2): 2 x 33.5 MB
  unsigned short* Abf = (unsigned short*)w; w += (size_t)B_ * N_ * N_ * 2;
  unsigned short* Tbf = (unsigned short*)w; w += (size_t)B_ * N_ * N_ * 2;
  const int use_pre = (ws_size >= (size_t)(w - (char*)d_ws)) ? 1 : 0;

  k_rowinv<<<B_ * N_, 256, 0, stream>>>(adj1, invr1);
  k_rowinv<<<B_ * N_, 256, 0, stream>>>(adj2, invr2);
  k_norm<<<(B_ * N_) / 8, 256, 0, stream>>>(z1, zn1);
  k_norm<<<(B_ * N_) / 8, 256, 0, stream>>>(z2, zn2);

  if (use_pre) {
    k_prep<<<B_ * 32 * 32, 256, 0, stream>>>(adj1, invr1, Abf, Tbf);
    k_score<true><<<B_ * (N_ / TM), 128, 0, stream>>>(adj1, invr1, zn1, Abf, Tbf, w1, i1);
    k_prep<<<B_ * 32 * 32, 256, 0, stream>>>(adj2, invr2, Abf, Tbf);
    k_score<true><<<B_ * (N_ / TM), 128, 0, stream>>>(adj2, invr2, zn2, Abf, Tbf, w2, i2);
  } else {
    k_score<false><<<B_ * (N_ / TM), 128, 0, stream>>>(adj1, invr1, zn1, 0, 0, w1, i1);
    k_score<false><<<B_ * (N_ / TM), 128, 0, stream>>>(adj2, invr2, zn2, 0, 0, w2, i2);
  }

  k_loss<<<B_ * (2 * N_) / TM, 128, 0, stream>>>(zn1, zn2, w1, i1, w2, i2, lossbuf);
  k_reduce<<<1, 256, 0, stream>>>(lossbuf, (float*)d_out);
}
